// DeepseekV2MoE_61220463837454
// MI455X (gfx1250) — compile-verified, matrix operand order checked by reading, not run on hardware
//
#include <hip/hip_runtime.h>
#include <hip/hip_bf16.h>
#include <math.h>

// ---------------- problem constants (match reference) ----------------
#define N_TOKENS   1024
#define HIDDEN     2048
#define N_EXPERTS  16
#define MOE_INTER  1024
#define TOP_K      6
#define N_GROUP    4
#define TOPK_GROUP 2
#define CAPACITY   1024
#define ROUTED_SCALING 2.5f

// ---------------- GEMM tiling ----------------
#define BM 64
#define BN 128
#define BK 32
#define LDPAD 40   // bf16 element stride in LDS; 80B rows -> conflict-free b128 frag reads

typedef __attribute__((ext_vector_type(16))) __bf16 v16bf;
typedef __attribute__((ext_vector_type(8)))  __bf16 v8bf;
typedef __attribute__((ext_vector_type(8)))  float  v8f;
typedef int v4i_vs __attribute__((vector_size(16)));   // matches builtin param pointee

#define WMMA_BF16(a, b, c) \
  __builtin_amdgcn_wmma_f32_16x16x32_bf16(false, (a), false, (b), (short)0, (c), false, false)

// Async global->LDS (CDNA5): raw 16B/lane copy tracked by ASYNCcnt.
#if defined(__has_builtin)
#if __has_builtin(__builtin_amdgcn_global_load_async_to_lds_b128)
#define HAVE_ASYNC_LDS 1
#endif
#endif

__device__ __forceinline__ void wait_async_then_none() {
#if defined(HAVE_ASYNC_LDS)
#if __has_builtin(__builtin_amdgcn_s_wait_asynccnt)
  __builtin_amdgcn_s_wait_asynccnt(0);
#else
  asm volatile("s_wait_asynccnt 0x0" ::: "memory");
#endif
#endif
}

// Load a 16-row x 32-K operand fragment from LDS laid out [row][K] with stride LDPAD.
// Per ISA 16-bit A layout: lanes 0-15 hold K {0..7,16..23}, lanes 16-31 hold K {8..15,24..31}.
// B tiles are stored N-major in LDS so the same loader produces the B fragment.
__device__ __forceinline__ v16bf frag_from_lds(const __bf16* base, int row_off, int lane) {
  const int r  = lane & 15;
  const int kh = (lane >> 4) << 3;            // 0 or 8
  const __bf16* p = base + (row_off + r) * LDPAD + kh;
  v8bf lo = *(const v8bf*)(p);                // K = kh + 0..7
  v8bf hi = *(const v8bf*)(p + 16);           // K = 16 + kh + 0..7
  v16bf f;
#pragma unroll
  for (int i = 0; i < 8; ++i) { f[i] = lo[i]; f[8 + i] = hi[i]; }
  return f;
}

__device__ __forceinline__ uint32_t pack2_bf16(float a, float b) {
  union { __bf16 h[2]; uint32_t u; } p;
  p.h[0] = (__bf16)a; p.h[1] = (__bf16)b;
  return p.u;
}

// ---------------- init: reset counts + pad slot tables ----------------
__global__ void init_kernel(int* __restrict__ cnt, int* __restrict__ slot_token,
                            float* __restrict__ slot_w) {
  int i = blockIdx.x * blockDim.x + threadIdx.x;
  if (i < N_EXPERTS) cnt[i] = 0;
  if (i < N_EXPERTS * CAPACITY) { slot_token[i] = N_TOKENS; slot_w[i] = 0.f; }
}

// ---------------- router: sigmoid scores, grouped top-k, atomic dispatch ----------------
__global__ void __launch_bounds__(256)
router_kernel(const float* __restrict__ x, const float* __restrict__ gate_w,
              const float* __restrict__ gate_bias, int* __restrict__ cnt,
              int* __restrict__ slot_token, float* __restrict__ slot_w) {
  const int t = blockIdx.x;
  __shared__ float part[N_EXPERTS][17];
  const int tid = threadIdx.x;
  const int e = tid >> 4;          // 16 threads per expert
  const int l = tid & 15;
  const float* xr = x + (size_t)t * HIDDEN;
  const float* wr = gate_w + (size_t)e * HIDDEN;
  float s = 0.f;
  for (int h = l; h < HIDDEN; h += 16) s += xr[h] * wr[h];
  part[e][l] = s;
  __syncthreads();
  if (tid < N_EXPERTS) {
    float tot = 0.f;
#pragma unroll
    for (int i = 0; i < 16; ++i) tot += part[tid][i];
    part[tid][16] = tot;
  }
  __syncthreads();
  if (tid == 0) {
    float scores[N_EXPERTS], sc[N_EXPERTS];
#pragma unroll
    for (int i = 0; i < N_EXPERTS; ++i) {
      float sg = 1.f / (1.f + __expf(-part[i][16]));
      scores[i] = sg;
      sc[i] = sg + gate_bias[i];
    }
    float gsc[N_GROUP];
#pragma unroll
    for (int g = 0; g < N_GROUP; ++g) {
      float m1 = -1e30f, m2 = -1e30f;
#pragma unroll
      for (int i = 0; i < N_EXPERTS / N_GROUP; ++i) {
        float v = sc[g * (N_EXPERTS / N_GROUP) + i];
        if (v > m1) { m2 = m1; m1 = v; } else if (v > m2) { m2 = v; }
      }
      gsc[g] = m1 + m2;
    }
    int g1 = 0;
    for (int g = 1; g < N_GROUP; ++g) if (gsc[g] > gsc[g1]) g1 = g;
    int g2 = -1;
    for (int g = 0; g < N_GROUP; ++g) {
      if (g == g1) continue;
      if (g2 < 0 || gsc[g] > gsc[g2]) g2 = g;
    }
    float masked[N_EXPERTS];
#pragma unroll
    for (int i = 0; i < N_EXPERTS; ++i) {
      int g = i >> 2;
      masked[i] = (g == g1 || g == g2) ? sc[i] : -1e30f;
    }
    int ids[TOP_K]; float w[TOP_K]; float wsum = 0.f;
    for (int k = 0; k < TOP_K; ++k) {
      int best = -1;
      for (int i = 0; i < N_EXPERTS; ++i) {
        bool used = false;
        for (int j = 0; j < k; ++j) if (ids[j] == i) used = true;
        if (used) continue;
        if (best < 0 || masked[i] > masked[best]) best = i;
      }
      ids[k] = best; w[k] = scores[best]; wsum += w[k];
    }
    const float inv = ROUTED_SCALING / wsum;
    for (int k = 0; k < TOP_K; ++k) {
      int ee = ids[k];
      int p = atomicAdd(&cnt[ee], 1);
      if (p < CAPACITY) {
        slot_token[ee * CAPACITY + p] = t;
        slot_w[ee * CAPACITY + p] = w[k] * inv;
      }
    }
  }
}

// ---------------- gate_up GEMM + fused SwiGLU (routed via gather, or shared direct) ----------------
// act[(e*rows_per_e + m)*Iw + n] = silu(Xe @ Wg)[m,n] * (Xe @ Wu)[m,n], bf16 output.
// 256 threads / 8 waves; BM=64 x BN=128 tile; wave computes 2x2 fragments for g and u.
__global__ void __launch_bounds__(256)
gateup_swiglu_kernel(const float* __restrict__ x, const float* __restrict__ wgu,
                     const int* __restrict__ slot_token, const int* __restrict__ cnt,
                     __bf16* __restrict__ act, int ldb, int Iw, int rows_per_e, int Kdim) {
  const int e  = blockIdx.z;
  const int m0 = blockIdx.y * BM;
  const int n0 = blockIdx.x * BN;
  if (cnt && m0 >= cnt[e]) return;   // uniform early-out over padded capacity

  const float* Bbase = wgu + (size_t)e * Kdim * ldb;

  __shared__ __attribute__((aligned(16))) __bf16 Alds[BM * LDPAD];
  __shared__ __attribute__((aligned(16))) __bf16 Bglds[BN * LDPAD];
  __shared__ __attribute__((aligned(16))) __bf16 Bulds[BN * LDPAD];

  const int tid  = threadIdx.x;
  const int lane = tid & 31;
  const int wid  = tid >> 5;
  const int ni0  = (wid & 3) * 2;       // fragment col pair: 0,2,4,6
  const int mi0  = (wid >> 2) * 2;      // fragment row pair: 0,2

  // A staging: 8 floats (2x float4) per thread, gather via slot table
  const int arow = tid >> 2;
  const int acol = (tid & 3) * 8;
  int tok = slot_token ? slot_token[e * rows_per_e + m0 + arow] : (m0 + arow);
  const bool avalid = (tok < N_TOKENS);
  const float* Arow = x + (size_t)tok * Kdim;

  // B staging: thread covers K-pair (bk2,bk2+1) x 8 columns; packed b32 LDS stores
  const int bk2 = (tid >> 4) * 2;       // 0,2,...,30
  const int bn  = (tid & 15) * 8;       // 0..120

  const v8f vzero = {0, 0, 0, 0, 0, 0, 0, 0};
  v8f accg[2][2] = {{vzero, vzero}, {vzero, vzero}};
  v8f accu[2][2] = {{vzero, vzero}, {vzero, vzero}};

  for (int k0 = 0; k0 < Kdim; k0 += BK) {
    { // stage A (fp32 -> bf16, gather with zero-padding)
      float4 f0 = {0, 0, 0, 0}, f1 = {0, 0, 0, 0};
      if (avalid) {
        f0 = *(const float4*)(Arow + k0 + acol);
        f1 = *(const float4*)(Arow + k0 + acol + 4);
      }
      v8bf av;
      av[0] = (__bf16)f0.x; av[1] = (__bf16)f0.y; av[2] = (__bf16)f0.z; av[3] = (__bf16)f0.w;
      av[4] = (__bf16)f1.x; av[5] = (__bf16)f1.y; av[6] = (__bf16)f1.z; av[7] = (__bf16)f1.w;
      *(v8bf*)(Alds + arow * LDPAD + acol) = av;
    }
    { // stage Bg / Bu transposed into LDS (N-major, two K packed per b32 store)
      const float* g0p = Bbase + (size_t)(k0 + bk2) * ldb + n0 + bn;
      const float* g1p = g0p + ldb;
      const float* u0p = g0p + Iw;
      const float* u1p = u0p + ldb;
      float4 ga = *(const float4*)(g0p);     float4 gb = *(const float4*)(g0p + 4);
      float4 gc = *(const float4*)(g1p);     float4 gd = *(const float4*)(g1p + 4);
      float4 ua = *(const float4*)(u0p);     float4 ub = *(const float4*)(u0p + 4);
      float4 uc = *(const float4*)(u1p);     float4 ud = *(const float4*)(u1p + 4);
      if (k0 + BK < Kdim) {
        __builtin_prefetch(g0p + (size_t)BK * ldb, 0, 3);
        __builtin_prefetch(u0p + (size_t)BK * ldb, 0, 3);
      }
      float gr0[8] = {ga.x, ga.y, ga.z, ga.w, gb.x, gb.y, gb.z, gb.w};
      float gr1[8] = {gc.x, gc.y, gc.z, gc.w, gd.x, gd.y, gd.z, gd.w};
      float ur0[8] = {ua.x, ua.y, ua.z, ua.w, ub.x, ub.y, ub.z, ub.w};
      float ur1[8] = {uc.x, uc.y, uc.z, uc.w, ud.x, ud.y, ud.z, ud.w};
#pragma unroll
      for (int j = 0; j < 8; ++j) {
        *(uint32_t*)(Bglds + (bn + j) * LDPAD + bk2) = pack2_bf16(gr0[j], gr1[j]);
        *(uint32_t*)(Bulds + (bn + j) * LDPAD + bk2) = pack2_bf16(ur0[j], ur1[j]);
      }
    }
    __syncthreads();
    v16bf afr[2], bgfr[2], bufr[2];
#pragma unroll
    for (int i = 0; i < 2; ++i) {
      afr[i]  = frag_from_lds(Alds,  (mi0 + i) * 16, lane);
      bgfr[i] = frag_from_lds(Bglds, (ni0 + i) * 16, lane);
      bufr[i] = frag_from_lds(Bulds, (ni0 + i) * 16, lane);
    }
#pragma unroll
    for (int fm = 0; fm < 2; ++fm)
#pragma unroll
      for (int fn = 0; fn < 2; ++fn) {
        accg[fm][fn] = WMMA_BF16(afr[fm], bgfr[fn], accg[fm][fn]);
        accu[fm][fn] = WMMA_BF16(afr[fm], bufr[fn], accu[fm][fn]);
      }
    __syncthreads();
  }

  // epilogue: SwiGLU, store bf16 activations
  const int rhalf = (lane >> 4) << 3;
#pragma unroll
  for (int fm = 0; fm < 2; ++fm) {
#pragma unroll
    for (int fn = 0; fn < 2; ++fn) {
      const int col = n0 + (ni0 + fn) * 16 + (lane & 15);
#pragma unroll
      for (int j = 0; j < 8; ++j) {
        int m = m0 + (mi0 + fm) * 16 + rhalf + j;
        float gv = accg[fm][fn][j];
        float av = (gv / (1.f + __expf(-gv))) * accu[fm][fn][j];
        act[(size_t)(e * rows_per_e + m) * Iw + col] = (__bf16)av;
      }
    }
  }
}

// ---------------- down GEMM: routed (weighted atomic scatter) or shared (plain store) ----------------
__global__ void __launch_bounds__(256)
down_kernel(const __bf16* __restrict__ act, const float* __restrict__ wd,
            const int* __restrict__ slot_token, const float* __restrict__ slot_w,
            const int* __restrict__ cnt, float* __restrict__ out,
            int Kdim, int ldb, int rows_per_e) {
  const int e  = blockIdx.z;
  const int m0 = blockIdx.y * BM;
  const int n0 = blockIdx.x * BN;
  if (cnt && m0 >= cnt[e]) return;

  const float* Bbase = wd + (size_t)e * Kdim * ldb;
  const __bf16* Abase = act + (size_t)e * rows_per_e * Kdim;

  __shared__ __attribute__((aligned(16))) __bf16 Alds[BM * LDPAD];
  __shared__ __attribute__((aligned(16))) __bf16 Blds[BN * LDPAD];

  const int tid  = threadIdx.x;
  const int lane = tid & 31;
  const int wid  = tid >> 5;
  const int ni0  = (wid & 3) * 2;
  const int mi0  = (wid >> 2) * 2;

  const int arow = tid >> 2;
  const int acol = (tid & 3) * 8;
  const int bk2 = (tid >> 4) * 2;
  const int bn  = (tid & 15) * 8;

  const v8f vzero = {0, 0, 0, 0, 0, 0, 0, 0};
  v8f acc[2][2] = {{vzero, vzero}, {vzero, vzero}};

  for (int k0 = 0; k0 < Kdim; k0 += BK) {
    { // stage A: bf16 source -> raw 16B/lane copy; async-to-LDS when available
      const __bf16* asrc = Abase + (size_t)(m0 + arow) * Kdim + k0 + acol;
      __bf16* adst = Alds + arow * LDPAD + acol;
#if defined(HAVE_ASYNC_LDS)
      __builtin_amdgcn_global_load_async_to_lds_b128(
          (__attribute__((address_space(1))) v4i_vs*)(asrc),
          (__attribute__((address_space(3))) v4i_vs*)(adst), 0, 0);
#else
      *(v8bf*)adst = *(const v8bf*)asrc;
#endif
    }
    { // stage B transposed into LDS (two K packed per b32 store)
      const float* b0p = Bbase + (size_t)(k0 + bk2) * ldb + n0 + bn;
      const float* b1p = b0p + ldb;
      float4 ba = *(const float4*)(b0p);  float4 bb = *(const float4*)(b0p + 4);
      float4 bc = *(const float4*)(b1p);  float4 bd = *(const float4*)(b1p + 4);
      if (k0 + BK < Kdim) __builtin_prefetch(b0p + (size_t)BK * ldb, 0, 3);
      float r0[8] = {ba.x, ba.y, ba.z, ba.w, bb.x, bb.y, bb.z, bb.w};
      float r1[8] = {bc.x, bc.y, bc.z, bc.w, bd.x, bd.y, bd.z, bd.w};
#pragma unroll
      for (int j = 0; j < 8; ++j)
        *(uint32_t*)(Blds + (bn + j) * LDPAD + bk2) = pack2_bf16(r0[j], r1[j]);
    }
    wait_async_then_none();   // ASYNCcnt==0 before barrier so A-tile is LDS-visible
    __syncthreads();
    v16bf afr[2], bfr[2];
#pragma unroll
    for (int i = 0; i < 2; ++i) {
      afr[i] = frag_from_lds(Alds, (mi0 + i) * 16, lane);
      bfr[i] = frag_from_lds(Blds, (ni0 + i) * 16, lane);
    }
#pragma unroll
    for (int fm = 0; fm < 2; ++fm)
#pragma unroll
      for (int fn = 0; fn < 2; ++fn)
        acc[fm][fn] = WMMA_BF16(afr[fm], bfr[fn], acc[fm][fn]);
    __syncthreads();
  }

  const int rhalf = (lane >> 4) << 3;
#pragma unroll
  for (int fm = 0; fm < 2; ++fm) {
#pragma unroll
    for (int fn = 0; fn < 2; ++fn) {
      const int col = n0 + (ni0 + fn) * 16 + (lane & 15);
#pragma unroll
      for (int j = 0; j < 8; ++j) {
        int m = m0 + (mi0 + fm) * 16 + rhalf + j;
        float v = acc[fm][fn][j];
        if (slot_token) {
          int idx = e * rows_per_e + m;
          int tokn = slot_token[idx];
          if (tokn < N_TOKENS)
            atomicAdd(out + (size_t)tokn * HIDDEN + col, v * slot_w[idx]);
        } else {
          out[(size_t)m * HIDDEN + col] = v;   // shared-expert path: sole writer
        }
      }
    }
  }
}

// ---------------- launch ----------------
extern "C" void kernel_launch(void* const* d_in, const int* in_sizes, int n_in,
                              void* d_out, int out_size, void* d_ws, size_t ws_size,
                              hipStream_t stream) {
  (void)in_sizes; (void)n_in; (void)out_size; (void)ws_size;
  const float* x      = (const float*)d_in[0];   // [1024, 2048]
  const float* gate_w = (const float*)d_in[1];   // [16, 2048]
  const float* gate_b = (const float*)d_in[2];   // [16]
  const float* w_gu   = (const float*)d_in[3];   // [16, 2048, 2048]
  const float* w_dn   = (const float*)d_in[4];   // [16, 1024, 2048]
  const float* s_gu   = (const float*)d_in[5];   // [2048, 4096]
  const float* s_dn   = (const float*)d_in[6];   // [2048, 2048]
  float* out = (float*)d_out;                    // [1024, 2048]

  char* ws = (char*)d_ws;
  int*    cnt        = (int*)ws;                                            // 64 B
  int*    slot_token = (int*)(ws + 256);                                    // 64 KB
  float*  slot_w     = (float*)(ws + 256 + N_EXPERTS * CAPACITY * 4);       // 64 KB
  __bf16* act_r      = (__bf16*)(ws + 256 + 2 * N_EXPERTS * CAPACITY * 4);  // 32 MB
  __bf16* act_s      = (__bf16*)((char*)act_r +
                        (size_t)N_EXPERTS * CAPACITY * MOE_INTER * 2);      // 4 MB

  init_kernel<<<(N_EXPERTS * CAPACITY + 255) / 256, 256, 0, stream>>>(cnt, slot_token, slot_w);
  router_kernel<<<N_TOKENS, 256, 0, stream>>>(x, gate_w, gate_b, cnt, slot_token, slot_w);

  // routed gate_up + SwiGLU: [E] x [C,2048] @ [2048,2048] -> act_r [E,C,1024] bf16
  gateup_swiglu_kernel<<<dim3(MOE_INTER / BN, CAPACITY / BM, N_EXPERTS), 256, 0, stream>>>(
      x, w_gu, slot_token, cnt, act_r, 2 * MOE_INTER, MOE_INTER, CAPACITY, HIDDEN);

  // shared gate_up + SwiGLU: [1024,2048] @ [2048,4096] -> act_s [1024,2048] bf16
  gateup_swiglu_kernel<<<dim3(2048 / BN, N_TOKENS / BM, 1), 256, 0, stream>>>(
      x, s_gu, nullptr, nullptr, act_s, 4096, 2048, N_TOKENS, HIDDEN);

  // shared down: act_s @ [2048,2048] -> out (plain stores; covers every element)
  down_kernel<<<dim3(HIDDEN / BN, N_TOKENS / BM, 1), 256, 0, stream>>>(
      act_s, s_dn, nullptr, nullptr, nullptr, out, 2048, HIDDEN, N_TOKENS);

  // routed down + weighted scatter-add into out
  down_kernel<<<dim3(HIDDEN / BN, CAPACITY / BM, N_EXPERTS), 256, 0, stream>>>(
      act_r, w_dn, slot_token, slot_w, cnt, out, MOE_INTER, HIDDEN, CAPACITY);
}
